// MultiSynonymsAttention_77257871720604
// MI455X (gfx1250) — matrix operand order repaired and checked by reading, not compiled, not run on hardware
//
#include <hip/hip_runtime.h>
#include <hip/hip_bf16.h>

typedef __attribute__((ext_vector_type(16))) _Float16 v16h;
typedef __attribute__((ext_vector_type(2)))  _Float16 v2h;
typedef __attribute__((ext_vector_type(8)))  float    v8f;

#define WMMA_F16(A, B, C) \
  __builtin_amdgcn_wmma_f32_16x16x32_f16(false, (A), false, (B), (short)0, (C), false, false)

// 16-bit A-matrix 16x32 layout (CDNA5 ISA):
// lane 0-15 hold M=0..15; element e -> VGPR v=e/2, lo=e&1;
// k = (v>=4 ? 16 : 0) + laneHalf*8 + (v&3)*2 + lo
__device__ __forceinline__ int a16_kbase(int half, int v) {
  return ((v >> 2) << 4) + (half << 3) + ((v & 3) << 1);
}

// ---------------------------------------------------------------------------
// Wt[k][n] = (f16) W[n][k]   (1024x1024 weight transpose + f16 convert)
// Gives B-operand loads (lane = N col, elements = K) consecutive n across lanes.
// ---------------------------------------------------------------------------
__global__ __launch_bounds__(256) void wtrans_kernel(const float* __restrict__ W,
                                                     _Float16* __restrict__ Wt) {
  int i = blockIdx.x * blockDim.x + threadIdx.x;   // 0 .. 1048576
  int k = i >> 10, n = i & 1023;
  Wt[(size_t)k * 1024 + n] = (_Float16)W[(size_t)n * 1024 + k];
}

// ---------------------------------------------------------------------------
// Wql[c][j] = sum_d ql[c][d] * WV_w[j][d]     (32 x 1024 output, fp32)
// ---------------------------------------------------------------------------
__global__ __launch_bounds__(256) void wql_kernel(const float* __restrict__ ql,
                                                  const float* __restrict__ WV,
                                                  float* __restrict__ Wql) {
  int i = blockIdx.x * blockDim.x + threadIdx.x;   // 0 .. 32768
  if (i >= 32 * 1024) return;
  int c = i >> 10, j = i & 1023;
  const float* a = ql + (size_t)c * 1024;
  const float* w = WV + (size_t)j * 1024;
  float s = 0.f;
#pragma unroll 8
  for (int d = 0; d < 1024; ++d) s += a[d] * w[d];
  Wql[i] = s;
}

// ---------------------------------------------------------------------------
// Kt[b][z][t][h] = tanh(H[b][t][z*64+h])   (f16, element-wise relayout)
// ---------------------------------------------------------------------------
__global__ __launch_bounds__(256) void kt_kernel(const float* __restrict__ H,
                                                 _Float16* __restrict__ Kt) {
  size_t i = (size_t)blockIdx.x * blockDim.x + threadIdx.x;  // 0 .. 16777216
  int col = (int)(i & 1023);
  int row = (int)(i >> 10);          // b*512 + t
  int b = row >> 9, t = row & 511;
  int z = col >> 6, h = col & 63;
  float v = tanhf(H[i]);
  Kt[(size_t)(((b * 16 + z) * 512 + t)) * 64 + h] = (_Float16)v;
}

// ---------------------------------------------------------------------------
// WMMA projection: one wave computes a 16x64 output strip; K=1024 (32 steps).
// A: 16x32 f32 tile of X (float2 loads + cvt, co-executes with WMMA pipe)
// B: 4 v16h gathered per element (stride 1024 over k) from pre-transposed f16 weights Wt[k][n]
// MODE 0: K-projection -> tanh -> WKtT[b][z][h][t]  (rows = b*512+t, cols = z*64+h)
// MODE 1: Q-projection ->        q_s [c][z][s][h]   (rows = c*16+s,  cols = z*64+h)
// blockIdx.x = rowTile*16 + colGroup   (colGroup = 64-wide strip)
// ---------------------------------------------------------------------------
template <int MODE>
__global__ __launch_bounds__(32) void proj_kernel(const float* __restrict__ X,
                                                  const _Float16* __restrict__ Wt,
                                                  _Float16* __restrict__ out) {
  int tile = blockIdx.x;
  int rt = tile >> 4;            // row tile
  int cg = tile & 15;            // col group (1024 / 64)
  int lane = threadIdx.x;
  int m = lane & 15, half = lane >> 4;

  v8f acc0 = {}, acc1 = {}, acc2 = {}, acc3 = {};
  const float* xrow = X + (size_t)(rt * 16 + m) * 1024;
#pragma unroll 2
  for (int ks = 0; ks < 32; ++ks) {
    int k0 = ks * 32;
    v16h a;
#pragma unroll
    for (int v = 0; v < 8; ++v) {
      int kb = a16_kbase(half, v);
      float2 f = *(const float2*)(xrow + k0 + kb);
      a[2 * v]     = (_Float16)f.x;
      a[2 * v + 1] = (_Float16)f.y;
    }
    const _Float16* wcol = Wt + (size_t)k0 * 1024 + cg * 64 + (lane & 15);  // lane&15 = N col
    v16h b0, b1, b2, b3;
#pragma unroll
    for (int v = 0; v < 8; ++v) {
      int kb = a16_kbase(half, v);
#pragma unroll
      for (int lo = 0; lo < 2; ++lo) {
        const _Float16* p = wcol + (size_t)(kb + lo) * 1024;
        b0[2 * v + lo] = p[0]; b1[2 * v + lo] = p[16]; b2[2 * v + lo] = p[32]; b3[2 * v + lo] = p[48];
      }
    }
    acc0 = WMMA_F16(a, b0, acc0);
    acc1 = WMMA_F16(a, b1, acc1);
    acc2 = WMMA_F16(a, b2, acc2);
    acc3 = WMMA_F16(a, b3, acc3);
  }
  // D layout: VGPR r -> M = r + 8*half, N = lane&15
  v8f* accs[4] = {&acc0, &acc1, &acc2, &acc3};
#pragma unroll
  for (int i = 0; i < 4; ++i) {
    v8f& acc = *accs[i];
#pragma unroll
    for (int r = 0; r < 8; ++r) {
      int row = rt * 16 + r + (half << 3);
      int col = cg * 64 + i * 16 + (lane & 15);
      float v = acc[r];
      if (MODE == 0) {
        int b = row >> 9, t = row & 511, z = col >> 6, h = col & 63;
        out[(size_t)(((b * 16 + z) * 64 + h)) * 512 + t] = (_Float16)tanhf(v);
      } else {
        int c = row >> 4, s = row & 15, z = col >> 6, h = col & 63;
        out[(size_t)(((c * 16 + z) * 16 + s)) * 64 + h] = (_Float16)v;
      }
    }
  }
}

// ---------------------------------------------------------------------------
// Attention: one wave per (b,c,z).  blockIdx.x = b*512 + c*16 + z
//   scores(16x512) = q_s(16x64) @ WKtT           (2 WMMA per t-tile)
//   softmax over t (in LDS, f16)
//   context(16x64) = attn(16x512) @ Kt           (16 k-steps x 4 h-tiles)
//   partial[b,c,z] = (1/16) * sum_{s,h} context[s,h] * Wql[c, z*64+h]
// ---------------------------------------------------------------------------
__global__ __launch_bounds__(32) void attn_kernel(const _Float16* __restrict__ WKtT,
                                                  const _Float16* __restrict__ Kt,
                                                  const _Float16* __restrict__ qs,
                                                  const float* __restrict__ Wql,
                                                  float* __restrict__ partial) {
  __shared__ _Float16 smem[16][512];   // 16 KB: scores, then attn weights (in-place)

  int id = blockIdx.x;
  int z = id & 15, c = (id >> 4) & 31, b = id >> 9;
  int lane = threadIdx.x;
  int m = lane & 15, half = lane >> 4;

  // q_s A fragments (16 x 64 -> two 16x32 fragments), per-lane paired loads
  v16h aq0, aq1;
  const _Float16* qrow = qs + (size_t)(((c * 16 + z) * 16 + m)) * 64;
#pragma unroll
  for (int v = 0; v < 8; ++v) {
    int kb = a16_kbase(half, v);
    v2h p0 = *(const v2h*)(qrow + kb);
    v2h p1 = *(const v2h*)(qrow + 32 + kb);
    aq0[2 * v] = p0[0]; aq0[2 * v + 1] = p0[1];
    aq1[2 * v] = p1[0]; aq1[2 * v + 1] = p1[1];
  }

  // ---- Phase 1: scores -> LDS (f16) ----
  const _Float16* wkbase = WKtT + (size_t)(b * 16 + z) * 64 * 512;
#pragma unroll 2
  for (int tt = 0; tt < 32; ++tt) {
    v8f acc = {};
    v16h b0;
#pragma unroll
    for (int v = 0; v < 8; ++v) {
      int kb = a16_kbase(half, v);
      b0[2 * v]     = wkbase[(size_t)kb * 512 + tt * 16 + m];
      b0[2 * v + 1] = wkbase[(size_t)(kb + 1) * 512 + tt * 16 + m];
    }
    acc = WMMA_F16(aq0, b0, acc);
    v16h b1;
#pragma unroll
    for (int v = 0; v < 8; ++v) {
      int kb = 32 + a16_kbase(half, v);
      b1[2 * v]     = wkbase[(size_t)kb * 512 + tt * 16 + m];
      b1[2 * v + 1] = wkbase[(size_t)(kb + 1) * 512 + tt * 16 + m];
    }
    acc = WMMA_F16(aq1, b1, acc);
#pragma unroll
    for (int r = 0; r < 8; ++r)
      smem[r + (half << 3)][tt * 16 + m] = (_Float16)acc[r];
  }
  __syncthreads();

  // ---- Phase 2: softmax over t, lane handles (row m, t-half) ----
  int t0 = half << 8;
  float mx = -1e30f;
  for (int t = 0; t < 256; ++t) mx = fmaxf(mx, (float)smem[m][t0 + t]);
  mx = fmaxf(mx, __shfl_xor(mx, 16));
  float sum = 0.f;
  for (int t = 0; t < 256; ++t) {
    float e = __expf((float)smem[m][t0 + t] - mx);
    sum += e;
    smem[m][t0 + t] = (_Float16)e;
  }
  sum += __shfl_xor(sum, 16);
  float inv = 1.0f / sum;
  for (int t = 0; t < 256; ++t)
    smem[m][t0 + t] = (_Float16)((float)smem[m][t0 + t] * inv);
  __syncthreads();

  // ---- Phase 3: context = attn @ Kt ----
  const _Float16* ktbase = Kt + (size_t)(b * 16 + z) * 512 * 64;
  v8f cacc0 = {}, cacc1 = {}, cacc2 = {}, cacc3 = {};
#pragma unroll 2
  for (int ks = 0; ks < 16; ++ks) {
    int tb = ks * 32;
    v16h a;
#pragma unroll
    for (int v = 0; v < 8; ++v) {
      int kb = a16_kbase(half, v);
      v2h p = *(const v2h*)(&smem[m][tb + kb]);
      a[2 * v] = p[0]; a[2 * v + 1] = p[1];
    }
    const _Float16* kcol = ktbase + (size_t)tb * 64 + m;  // m = N col (h)
    v16h b0, b1, b2, b3;
#pragma unroll
    for (int v = 0; v < 8; ++v) {
      int kb = a16_kbase(half, v);
#pragma unroll
      for (int lo = 0; lo < 2; ++lo) {
        const _Float16* p = kcol + (size_t)(kb + lo) * 64;
        b0[2 * v + lo] = p[0]; b1[2 * v + lo] = p[16]; b2[2 * v + lo] = p[32]; b3[2 * v + lo] = p[48];
      }
    }
    cacc0 = WMMA_F16(a, b0, cacc0);
    cacc1 = WMMA_F16(a, b1, cacc1);
    cacc2 = WMMA_F16(a, b2, cacc2);
    cacc3 = WMMA_F16(a, b3, cacc3);
  }

  // ---- Phase 4: mean over s, dot with Wql, reduce ----
  const float* wq = Wql + (size_t)c * 1024 + z * 64;
  float res = 0.f;
  v8f* caccs[4] = {&cacc0, &cacc1, &cacc2, &cacc3};
#pragma unroll
  for (int ht = 0; ht < 4; ++ht) {
    v8f& cc = *caccs[ht];
    float cs = 0.f;
#pragma unroll
    for (int r = 0; r < 8; ++r) cs += cc[r];
    cs += __shfl_xor(cs, 16);              // sum over all 16 rows (s)
    float p = cs * wq[ht * 16 + (lane & 15)];
    p += __shfl_xor(p, 8);                 // reduce 16-lane group over h
    p += __shfl_xor(p, 4);
    p += __shfl_xor(p, 2);
    p += __shfl_xor(p, 1);
    res += p;
  }
  if (lane == 0) partial[id] = res * (1.0f / 16.0f);
}

// ---------------------------------------------------------------------------
// out[b][c] = sum_z partial[b*512 + c*16 + z]   (fixed order -> deterministic)
// ---------------------------------------------------------------------------
__global__ __launch_bounds__(256) void final_kernel(const float* __restrict__ partial,
                                                    float* __restrict__ out) {
  int i = blockIdx.x * blockDim.x + threadIdx.x;   // 0..1024
  if (i >= 1024) return;
  float s = 0.f;
#pragma unroll
  for (int z = 0; z < 16; ++z) s += partial[i * 16 + z];
  out[i] = s;
}

extern "C" void kernel_launch(void* const* d_in, const int* in_sizes, int n_in,
                              void* d_out, int out_size, void* d_ws, size_t ws_size,
                              hipStream_t stream) {
  const float* Q    = (const float*)d_in[0];  // (32, 16, 1024)
  const float* H    = (const float*)d_in[1];  // (32, 512, 1024)
  const float* ql   = (const float*)d_in[2];  // (32, 1024)
  const float* WQ_w = (const float*)d_in[3];  // (1024, 1024)
  const float* WK_w = (const float*)d_in[4];  // (1024, 1024)
  const float* WV_w = (const float*)d_in[5];  // (1024, 1024)
  float* out = (float*)d_out;                 // (32, 32)

  char* ws = (char*)d_ws;
  size_t off = 0;
  _Float16* WKtT  = (_Float16*)(ws + off); off += 33554432;  // [32][16][64][512] f16 = 32 MB
  _Float16* Kt    = (_Float16*)(ws + off); off += 33554432;  // [32][16][512][64] f16 = 32 MB
  _Float16* qsbuf = (_Float16*)(ws + off); off += 1048576;   // [32][16][16][64] f16 = 1 MB
  _Float16* WKwT  = (_Float16*)(ws + off); off += 2097152;   // 1024x1024 f16 transposed = 2 MB
  _Float16* WQwT  = (_Float16*)(ws + off); off += 2097152;   // 1024x1024 f16 transposed = 2 MB
  float*    Wql   = (float*)   (ws + off); off += 131072;    // 32x1024 f32
  float*  partial = (float*)   (ws + off); off += 65536;     // 16384 f32

  // weight transposes (f32 -> f16, [k][n] layout for contiguous B loads)
  wtrans_kernel<<<4096, 256, 0, stream>>>(WK_w, WKwT);
  wtrans_kernel<<<4096, 256, 0, stream>>>(WQ_w, WQwT);

  // small projections / relayouts
  wql_kernel<<<128, 256, 0, stream>>>(ql, WV_w, Wql);
  kt_kernel<<<65536, 256, 0, stream>>>(H, Kt);

  // WMMA projections: one wave per 16x64 output strip
  proj_kernel<0><<<(16384 / 16) * 16, 32, 0, stream>>>(H, WKwT, WKtT);   // 16384 strips
  proj_kernel<1><<<(512 / 16) * 16, 32, 0, stream>>>(Q, WQwT, qsbuf);    // 512 strips

  // attention: one wave per (b, c, z)
  attn_kernel<<<32 * 32 * 16, 32, 0, stream>>>(WKtT, Kt, qsbuf, Wql, partial);

  // deterministic final reduction over z
  final_kernel<<<4, 256, 0, stream>>>(partial, out);
}
